// CrossAttention_50139448213962
// MI455X (gfx1250) — matrix operand order repaired and checked by reading, not compiled, not run on hardware
//
#include <hip/hip_runtime.h>
#include <hip/hip_bf16.h>

typedef _Float16 half_t;
typedef __attribute__((ext_vector_type(16))) _Float16 v16h;
typedef __attribute__((ext_vector_type(8)))  float    v8f;
typedef __attribute__((ext_vector_type(4)))  _Float16 h4;

#define B_   4
#define S1_  4096
#define S2_  256
#define QD_  1408
#define KVD_ 2816
#define KD_  1024
#define H_   16
#define D_   88
#define DP_  96
#define SCALE_Q 0.10660035817780521f  // 1/sqrt(88)

// ---------------------------------------------------------------------------
// CDNA5 async global->LDS copy (ASYNCcnt-tracked; CDNA5 ISA).
// VDST VGPR holds the per-lane LDS byte offset; GV addressing from a VGPR pair.
// Generic pointers to __shared__ carry the LDS offset in their low 32 bits
// ---------------------------------------------------------------------------
__device__ __forceinline__ unsigned lds_off(const void* p) {
  return (unsigned)(uintptr_t)p;
}
__device__ __forceinline__ void async_ld_b128(unsigned dst_lds, const half_t* src) {
  asm volatile("global_load_async_to_lds_b128 %0, %1, off"
               :: "v"(dst_lds), "v"(src)
               : "memory");
}
__device__ __forceinline__ void wait_async() {
  asm volatile("s_wait_asynccnt 0x0" ::: "memory");
}

// ---------------------------------------------------------------------------
// WMMA fragment loaders (per CDNA5 ISA 7.12.2 VGPR layouts, wave32)
// A (16x32 f16): lanes 0-15 -> M=lane, K in {0..7,16..23}; lanes 16-31 -> +8 K offset
// B (32x16 f16): lanes 0-15 -> N=lane, K in {0..7,16..23}; lanes 16-31 -> N=lane-16, +8 K offset
// ---------------------------------------------------------------------------
__device__ __forceinline__ v16h ldfragA(const half_t* base, int row, int stride,
                                        int kofs, int lane) {
  const int m  = row + (lane & 15);
  const int k0 = kofs + ((lane >> 4) << 3);
  union { v16h v; uint4 u[2]; } f;
  f.u[0] = *(const uint4*)(base + m * stride + k0);
  f.u[1] = *(const uint4*)(base + m * stride + k0 + 16);
  return f.v;
}

__device__ __forceinline__ v16h ldfragB(const half_t* base, int row, int stride,
                                        int kofs, int lane) {
  const int n  = row + (lane & 15);
  const int k0 = kofs + ((lane >> 4) << 3);
  union { v16h v; uint4 u[2]; } f;
  f.u[0] = *(const uint4*)(base + n * stride + k0);
  f.u[1] = *(const uint4*)(base + n * stride + k0 + 16);
  return f.v;
}

__device__ __forceinline__ v8f wmma_f16(v16h a, v16h b, v8f c) {
  return __builtin_amdgcn_wmma_f32_16x16x32_f16(false, a, false, b, (short)0, c,
                                                false, false);
}

// ---------------------------------------------------------------------------
// f32 -> f16 convert (4-wide)
// ---------------------------------------------------------------------------
__global__ void __launch_bounds__(256) cvt_f16_kernel(const float* __restrict__ in,
                                                      half_t* __restrict__ out, int n4) {
  const int i = blockIdx.x * 256 + threadIdx.x;
  if (i >= n4) return;
  const float4 v = ((const float4*)in)[i];
  h4 o;
  o.x = (half_t)v.x; o.y = (half_t)v.y; o.z = (half_t)v.z; o.w = (half_t)v.w;
  ((h4*)out)[i] = o;
}

// ---------------------------------------------------------------------------
// Generic f16 GEMM: C[M,N] = A[M,K] * W[N,K]^T + bias[N]  (C in f32)
// Block tile 128x128, 8 waves (4 in M x 2 in N), wave tile 32x64.
// Double-buffered LDS filled by async global->LDS; one barrier per K-step.
// ---------------------------------------------------------------------------
__global__ void __launch_bounds__(256) gemm_f16_nt(
    const half_t* __restrict__ A, const half_t* __restrict__ W,
    const float* __restrict__ bias, float* __restrict__ C,
    int M, int N, int K) {
  __shared__ half_t aS[2][128 * 40];
  __shared__ half_t bS[2][128 * 40];
  const int tid  = threadIdx.x;
  const int lane = tid & 31;
  const int wave = tid >> 5;
  const int tileM = blockIdx.y * 128;
  const int tileN = blockIdx.x * 128;
  const int wm = (wave >> 1) * 32;
  const int wn = (wave & 1) * 64;

  const int lrow = tid >> 1;          // 0..127
  const int lcol = (tid & 1) * 16;    // 0 or 16 halfs
  const half_t* Ag = A + (size_t)(tileM + lrow) * K + lcol;
  const half_t* Wg = W + (size_t)(tileN + lrow) * K + lcol;

  v8f acc[2][4] = {};

  auto stage = [&](int buf, int kt) {
    const half_t* An = Ag + (size_t)kt * 32;
    const half_t* Wn = Wg + (size_t)kt * 32;
    async_ld_b128(lds_off(aS[buf] + lrow * 40 + lcol),     An);
    async_ld_b128(lds_off(aS[buf] + lrow * 40 + lcol + 8), An + 8);
    async_ld_b128(lds_off(bS[buf] + lrow * 40 + lcol),     Wn);
    async_ld_b128(lds_off(bS[buf] + lrow * 40 + lcol + 8), Wn + 8);
  };

  stage(0, 0);
  wait_async();
  __syncthreads();

  int buf = 0;
  const int nk = K >> 5;
  for (int kt = 0; kt < nk; ++kt) {
    if (kt + 1 < nk) stage(buf ^ 1, kt + 1);   // fill next buffer asynchronously
    v16h af[2], bf[4];
#pragma unroll
    for (int i = 0; i < 2; ++i) af[i] = ldfragA(aS[buf], wm + i * 16, 40, 0, lane);
#pragma unroll
    for (int j = 0; j < 4; ++j) bf[j] = ldfragB(bS[buf], wn + j * 16, 40, 0, lane);
#pragma unroll
    for (int i = 0; i < 2; ++i)
#pragma unroll
      for (int j = 0; j < 4; ++j)
        acc[i][j] = wmma_f16(af[i], bf[j], acc[i][j]);
    wait_async();      // this wave's fills of buf^1 are in LDS
    __syncthreads();   // all waves done reading buf and done filling buf^1
    buf ^= 1;
  }

  // Epilogue: C-fragment mapping: m = r + 8*(lane>>4), n = lane&15
#pragma unroll
  for (int j = 0; j < 4; ++j) {
    const int n  = tileN + wn + j * 16 + (lane & 15);
    const float bv = bias[n];
#pragma unroll
    for (int i = 0; i < 2; ++i) {
      const int m0 = tileM + wm + i * 16 + ((lane >> 4) << 3);
#pragma unroll
      for (int r = 0; r < 8; ++r)
        C[(size_t)(m0 + r) * N + n] = acc[i][j][r] + bv;
    }
  }
}

// ---------------------------------------------------------------------------
// Q path: LayerNorm(D=88) + RoPE + *scale, f32 -> padded f16 [b,h,s,96]
// One wave per (b,s,h) row. Lane holds d = {lane, lane+32, lane+64(<88)}.
// RoPE partner (d^1) fetched via __shfl_xor(.,1).
// ---------------------------------------------------------------------------
__global__ void __launch_bounds__(256) lnrope_q_kernel(
    const float* __restrict__ q, const float* __restrict__ cosd,
    const float* __restrict__ sind, const float* __restrict__ gw,
    const float* __restrict__ gb, half_t* __restrict__ qp) {
  const int lane = threadIdx.x & 31;
  const int warp = threadIdx.x >> 5;
  const int gh = blockIdx.x * 8 + warp;     // [0, B*S1*H)
  const int bs = gh >> 4;
  const int h  = gh & 15;
  const int s  = bs & (S1_ - 1);
  const int b  = bs >> 12;
  const float* row = q + (size_t)bs * QD_ + h * D_;
  const bool v2 = lane < 24;
  const int d0 = lane, d1 = lane + 32, d2 = lane + 64;
  const float x0 = row[d0];
  const float x1 = row[d1];
  const float x2 = v2 ? row[d2] : 0.f;
  float sum = x0 + x1 + x2;
#pragma unroll
  for (int o = 16; o > 0; o >>= 1) sum += __shfl_xor(sum, o, 32);
  const float mu = sum * (1.f / 88.f);
  const float e0 = x0 - mu, e1 = x1 - mu, e2 = v2 ? (x2 - mu) : 0.f;
  float ss = e0 * e0 + e1 * e1 + e2 * e2;
#pragma unroll
  for (int o = 16; o > 0; o >>= 1) ss += __shfl_xor(ss, o, 32);
  const float rstd = rsqrtf(ss * (1.f / 88.f) + 1e-6f);
  const float n0 = e0 * rstd * gw[d0] + gb[d0];
  const float n1 = e1 * rstd * gw[d1] + gb[d1];
  const float n2 = v2 ? (e2 * rstd * gw[d2] + gb[d2]) : 0.f;
  const float p0 = __shfl_xor(n0, 1, 32);
  const float p1 = __shfl_xor(n1, 1, 32);
  const float p2 = __shfl_xor(n2, 1, 32);
  const float sg = (lane & 1) ? 1.f : -1.f;   // rotate_half: even d uses -x[d+1]
  const float* cr = cosd + (size_t)s * D_;
  const float* sr = sind + (size_t)s * D_;
  const float r0 = n0 * cr[d0] + sg * p0 * sr[d0];
  const float r1 = n1 * cr[d1] + sg * p1 * sr[d1];
  float r2 = 0.f;
  if (v2) r2 = n2 * cr[d2] + sg * p2 * sr[d2];
  half_t* out = qp + (((size_t)(b * H_ + h) * S1_) + s) * DP_;
  out[d0] = (half_t)(r0 * SCALE_Q);
  out[d1] = (half_t)(r1 * SCALE_Q);
  out[d2] = v2 ? (half_t)(r2 * SCALE_Q) : (half_t)0.f;  // zero pad 88..95
}

// ---------------------------------------------------------------------------
// K path: LayerNorm -> kp [b,h,s2,96] (zero padded).
// V path: transpose into vt [b,h,d(96),s2] for the PV GEMM B-matrix.
// ---------------------------------------------------------------------------
__global__ void __launch_bounds__(256) lnk_v_kernel(
    const float* __restrict__ kv, const float* __restrict__ gw,
    const float* __restrict__ gb, half_t* __restrict__ kp,
    half_t* __restrict__ vt) {
  const int lane = threadIdx.x & 31;
  const int warp = threadIdx.x >> 5;
  const int gh = blockIdx.x * 8 + warp;     // [0, B*S2*H)
  const int bs = gh >> 4;
  const int h  = gh & 15;
  const int s  = bs & (S2_ - 1);
  const int b  = bs >> 8;
  const bool v2 = lane < 24;
  const int d0 = lane, d1 = lane + 32, d2 = lane + 64;
  const float* krow = kv + (size_t)bs * KVD_ + h * D_;
  const float x0 = krow[d0];
  const float x1 = krow[d1];
  const float x2 = v2 ? krow[d2] : 0.f;
  float sum = x0 + x1 + x2;
#pragma unroll
  for (int o = 16; o > 0; o >>= 1) sum += __shfl_xor(sum, o, 32);
  const float mu = sum * (1.f / 88.f);
  const float e0 = x0 - mu, e1 = x1 - mu, e2 = v2 ? (x2 - mu) : 0.f;
  float ss = e0 * e0 + e1 * e1 + e2 * e2;
#pragma unroll
  for (int o = 16; o > 0; o >>= 1) ss += __shfl_xor(ss, o, 32);
  const float rstd = rsqrtf(ss * (1.f / 88.f) + 1e-6f);
  const float n0 = e0 * rstd * gw[d0] + gb[d0];
  const float n1 = e1 * rstd * gw[d1] + gb[d1];
  const float n2 = v2 ? (e2 * rstd * gw[d2] + gb[d2]) : 0.f;
  half_t* kout = kp + (((size_t)(b * H_ + h) * S2_) + s) * DP_;
  kout[d0] = (half_t)n0;
  kout[d1] = (half_t)n1;
  kout[d2] = v2 ? (half_t)n2 : (half_t)0.f;
  // V transpose (d-major); pad rows 88..95 zeroed
  const float* vrow = krow + QD_;
  half_t* vbase = vt + (size_t)(b * H_ + h) * DP_ * S2_;
  vbase[(size_t)d0 * S2_ + s] = (half_t)vrow[d0];
  vbase[(size_t)d1 * S2_ + s] = (half_t)vrow[d1];
  vbase[(size_t)d2 * S2_ + s] = v2 ? (half_t)vrow[d2] : (half_t)0.f;
}

// ---------------------------------------------------------------------------
// scores+softmax: per (b,h, 128 q-rows). Each wave owns 16 full rows of the
// 128x256 score tile in 16 WMMA accumulators; softmax done in registers with
// intra-half-wave shuffles (C-fragment halves are independent 16-lane groups).
// Tiles staged with async global->LDS. Writes P (f16) [b,h,s1,256].
// ---------------------------------------------------------------------------
__global__ void __launch_bounds__(256) attn_scores_kernel(
    const half_t* __restrict__ qp, const half_t* __restrict__ kp,
    half_t* __restrict__ p) {
  __shared__ half_t qS[128 * 104];
  __shared__ half_t kS[128 * 104];
  const int tid = threadIdx.x, lane = tid & 31, wave = tid >> 5;
  const int mt = blockIdx.x, h = blockIdx.y, b = blockIdx.z;
  const size_t bh = (size_t)b * H_ + h;
  const half_t* qg = qp + (bh * S1_ + (size_t)mt * 128) * DP_;
  const half_t* kg = kp + bh * S2_ * DP_;
  for (int i = tid; i < 128 * 12; i += 256) {
    const int r = i / 12, c = i - r * 12;
    async_ld_b128(lds_off(qS + r * 104 + c * 8), qg + (size_t)r * DP_ + c * 8);
  }
  v8f acc[16] = {};
#pragma unroll
  for (int hk = 0; hk < 2; ++hk) {
    __syncthreads();   // all waves done reading kS from previous pass
    for (int i = tid; i < 128 * 12; i += 256) {
      const int r = i / 12, c = i - r * 12;
      async_ld_b128(lds_off(kS + r * 104 + c * 8),
                    kg + (size_t)(hk * 128 + r) * DP_ + c * 8);
    }
    wait_async();      // covers qS fills too on the first pass
    __syncthreads();
#pragma unroll
    for (int kt = 0; kt < 3; ++kt) {
      const v16h af = ldfragA(qS, wave * 16, 104, kt * 32, lane);
#pragma unroll
      for (int j = 0; j < 8; ++j) {
        const v16h bf = ldfragB(kS, j * 16, 104, kt * 32, lane);
        acc[hk * 8 + j] = wmma_f16(af, bf, acc[hk * 8 + j]);
      }
    }
  }
  // softmax over 256 cols; rows r (lanes 0-15) and r+8 (lanes 16-31) in parallel
  const int m0 = (lane >> 4) << 3;
#pragma unroll
  for (int r = 0; r < 8; ++r) {
    float pm = -3.0e38f;
#pragma unroll
    for (int j = 0; j < 16; ++j) pm = fmaxf(pm, acc[j][r]);
#pragma unroll
    for (int o = 1; o < 16; o <<= 1) pm = fmaxf(pm, __shfl_xor(pm, o, 32));
    float sum = 0.f;
#pragma unroll
    for (int j = 0; j < 16; ++j) {
      const float e = __expf(acc[j][r] - pm);
      acc[j][r] = e;
      sum += e;
    }
#pragma unroll
    for (int o = 1; o < 16; o <<= 1) sum += __shfl_xor(sum, o, 32);
    const float inv = 1.f / sum;
    const size_t row = bh * S1_ + (size_t)mt * 128 + wave * 16 + r + m0;
    half_t* pr = p + row * 256 + (lane & 15);
#pragma unroll
    for (int j = 0; j < 16; ++j) pr[j * 16] = (half_t)(acc[j][r] * inv);
  }
}

// ---------------------------------------------------------------------------
// ctx = P(128x256) x V(256x96), batched over (b,h). Async double-buffered LDS.
// Writes ctx f16 [b,s1,h*88+d].
// ---------------------------------------------------------------------------
__global__ void __launch_bounds__(256) attn_pv_kernel(
    const half_t* __restrict__ p, const half_t* __restrict__ vt,
    half_t* __restrict__ ctx) {
  __shared__ half_t aS[2][128 * 40];
  __shared__ half_t bS[2][96 * 40];
  const int tid = threadIdx.x, lane = tid & 31, wave = tid >> 5;
  const int mt = blockIdx.x;
  const int bh = blockIdx.y;
  const int b = bh >> 4, h = bh & 15;
  const half_t* Ag = p + ((size_t)bh * S1_ + (size_t)mt * 128) * 256;
  const half_t* Bg = vt + (size_t)bh * DP_ * 256;
  const int wm = (wave >> 1) * 32;
  const int wn = (wave & 1) * 48;
  v8f acc[2][3] = {};
  // A: 512 16B-chunks (2/thread); B: 384 chunks (threads<128 take 2)
  const int a0r = tid >> 2,          a0c = (tid & 3) << 3;
  const int a1r = (tid + 256) >> 2,  a1c = ((tid + 256) & 3) << 3;
  const int b0r = tid >> 2,          b0c = (tid & 3) << 3;
  const int b1r = (tid + 256) >> 2,  b1c = ((tid + 256) & 3) << 3;
  const bool hb = tid < 128;

  auto stage = [&](int buf, int kt) {
    const int ko = kt * 32;
    async_ld_b128(lds_off(aS[buf] + a0r * 40 + a0c), Ag + (size_t)a0r * 256 + ko + a0c);
    async_ld_b128(lds_off(aS[buf] + a1r * 40 + a1c), Ag + (size_t)a1r * 256 + ko + a1c);
    async_ld_b128(lds_off(bS[buf] + b0r * 40 + b0c), Bg + (size_t)b0r * 256 + ko + b0c);
    if (hb)
      async_ld_b128(lds_off(bS[buf] + b1r * 40 + b1c), Bg + (size_t)b1r * 256 + ko + b1c);
  };

  stage(0, 0);
  wait_async();
  __syncthreads();
  int buf = 0;
  for (int kt = 0; kt < 8; ++kt) {
    if (kt < 7) stage(buf ^ 1, kt + 1);
    v16h af[2], bf[3];
#pragma unroll
    for (int i = 0; i < 2; ++i) af[i] = ldfragA(aS[buf], wm + i * 16, 40, 0, lane);
#pragma unroll
    for (int j = 0; j < 3; ++j) bf[j] = ldfragB(bS[buf], wn + j * 16, 40, 0, lane);
#pragma unroll
    for (int i = 0; i < 2; ++i)
#pragma unroll
      for (int j = 0; j < 3; ++j)
        acc[i][j] = wmma_f16(af[i], bf[j], acc[i][j]);
    wait_async();
    __syncthreads();
    buf ^= 1;
  }
#pragma unroll
  for (int j = 0; j < 3; ++j) {
    const int d = wn + j * 16 + (lane & 15);
    if (d < D_) {
#pragma unroll
      for (int i = 0; i < 2; ++i) {
        const int mrow = mt * 128 + wm + i * 16 + ((lane >> 4) << 3);
#pragma unroll
        for (int r = 0; r < 8; ++r)
          ctx[((size_t)b * S1_ + mrow + r) * QD_ + h * D_ + d] = (half_t)acc[i][j][r];
      }
    }
  }
}

// ---------------------------------------------------------------------------
// host side
// ---------------------------------------------------------------------------
extern "C" void kernel_launch(void* const* d_in, const int* in_sizes, int n_in,
                              void* d_out, int out_size, void* d_ws, size_t ws_size,
                              hipStream_t stream) {
  (void)in_sizes; (void)n_in; (void)out_size; (void)ws_size;
  const float* x    = (const float*)d_in[0];
  const float* y    = (const float*)d_in[1];
  const float* cosd = (const float*)d_in[2];
  const float* sind = (const float*)d_in[3];
  const float* Wq   = (const float*)d_in[4];
  const float* bq   = (const float*)d_in[5];
  const float* Wkv  = (const float*)d_in[6];
  const float* bkv  = (const float*)d_in[7];
  const float* qnw  = (const float*)d_in[8];
  const float* qnb  = (const float*)d_in[9];
  const float* knw  = (const float*)d_in[10];
  const float* knb  = (const float*)d_in[11];
  const float* Wout = (const float*)d_in[12];
  const float* bout = (const float*)d_in[13];

  constexpr size_t SZ_X16  = (size_t)B_ * S1_ * QD_ * 2;   // 46,137,344
  constexpr size_t SZ_Y16  = (size_t)B_ * S2_ * KD_ * 2;
  constexpr size_t SZ_WQ   = (size_t)QD_ * QD_ * 2;
  constexpr size_t SZ_WKV  = (size_t)KVD_ * KD_ * 2;
  constexpr size_t SZ_WO   = (size_t)QD_ * QD_ * 2;
  constexpr size_t SZ_KV32 = (size_t)B_ * S2_ * KVD_ * 4;
  constexpr size_t SZ_Q32  = (size_t)B_ * S1_ * QD_ * 4;
  constexpr size_t SZ_QP   = (size_t)B_ * H_ * S1_ * DP_ * 2;
  constexpr size_t SZ_KP   = (size_t)B_ * H_ * S2_ * DP_ * 2;

  constexpr size_t OFF_X16  = 0;
  constexpr size_t OFF_Y16  = OFF_X16 + SZ_X16;
  constexpr size_t OFF_WQ   = OFF_Y16 + SZ_Y16;
  constexpr size_t OFF_WKV  = OFF_WQ + SZ_WQ;
  constexpr size_t OFF_WO   = OFF_WKV + SZ_WKV;
  constexpr size_t OFF_KV32 = OFF_WO + SZ_WO;
  constexpr size_t OFF_Q32  = OFF_KV32 + SZ_KV32;
  constexpr size_t OFF_QP   = OFF_Q32 + SZ_Q32;
  constexpr size_t OFF_KP   = OFF_QP + SZ_QP;
  constexpr size_t OFF_VT   = OFF_KP + SZ_KP;

  char* ws = (char*)d_ws;
  half_t* x16  = (half_t*)(ws + OFF_X16);
  half_t* y16  = (half_t*)(ws + OFF_Y16);
  half_t* wq16 = (half_t*)(ws + OFF_WQ);
  half_t* wkv16= (half_t*)(ws + OFF_WKV);
  half_t* wo16 = (half_t*)(ws + OFF_WO);
  float*  kv32 = (float*)(ws + OFF_KV32);
  float*  q32  = (float*)(ws + OFF_Q32);
  half_t* qp16 = (half_t*)(ws + OFF_QP);
  half_t* kp16 = (half_t*)(ws + OFF_KP);
  half_t* vt16 = (half_t*)(ws + OFF_VT);
  half_t* p16  = (half_t*)(ws + OFF_X16);  // alias: x16 dead after Q-proj GEMM
  half_t* ctx  = (half_t*)(ws + OFF_Q32);  // alias: q32 dead after lnrope

  auto cvt = [&](const float* in, half_t* out, size_t n) {
    const int n4 = (int)(n / 4);
    cvt_f16_kernel<<<dim3((n4 + 255) / 256), dim3(256), 0, stream>>>(in, out, n4);
  };
  cvt(x,    x16,  (size_t)B_ * S1_ * QD_);
  cvt(y,    y16,  (size_t)B_ * S2_ * KD_);
  cvt(Wq,   wq16, (size_t)QD_ * QD_);
  cvt(Wkv,  wkv16,(size_t)KVD_ * KD_);
  cvt(Wout, wo16, (size_t)QD_ * QD_);

  // Q projection: [16384,1408] = x16 * Wq^T + bq
  gemm_f16_nt<<<dim3(QD_ / 128, (B_ * S1_) / 128), dim3(256), 0, stream>>>(
      x16, wq16, bq, q32, B_ * S1_, QD_, QD_);
  // KV projection: [1024,2816] = y16 * Wkv^T + bkv
  gemm_f16_nt<<<dim3(KVD_ / 128, (B_ * S2_) / 128), dim3(256), 0, stream>>>(
      y16, wkv16, bkv, kv32, B_ * S2_, KVD_, KD_);
  // LN + RoPE + scale on q; LN on k; v transpose
  lnrope_q_kernel<<<dim3((B_ * S1_ * H_) / 8), dim3(256), 0, stream>>>(
      q32, cosd, sind, qnw, qnb, qp16);
  lnk_v_kernel<<<dim3((B_ * S2_ * H_) / 8), dim3(256), 0, stream>>>(
      kv32, knw, knb, kp16, vt16);
  // attention
  attn_scores_kernel<<<dim3(S1_ / 128, H_, B_), dim3(256), 0, stream>>>(
      qp16, kp16, p16);
  attn_pv_kernel<<<dim3(S1_ / 128, B_ * H_), dim3(256), 0, stream>>>(
      p16, vt16, ctx);
  // output projection straight into d_out (f32)
  gemm_f16_nt<<<dim3(QD_ / 128, (B_ * S1_) / 128), dim3(256), 0, stream>>>(
      ctx, wo16, bout, (float*)d_out, B_ * S1_, QD_, QD_);
}
